// RecurrentODESolver_25460566130815
// MI455X (gfx1250) — matrix operand order repaired and checked by reading, not compiled, not run on hardware
//
#include <hip/hip_runtime.h>
#include <math.h>

typedef __bf16 bf16_t;
typedef __attribute__((ext_vector_type(16))) __bf16 v16bf;
typedef __attribute__((ext_vector_type(8)))  __bf16 v8bf;
typedef __attribute__((ext_vector_type(8)))  float  v8f;

#define WSTRIDE 264   // bf16 elements per LDS row (256 + 8 pad -> conflict-free b128 reads)
#define NSTEPS  40
#define HSTEP   0.025f

#if defined(__gfx1250__) && __has_builtin(__builtin_amdgcn_tanhf)
#define TANHF(x) __builtin_amdgcn_tanhf(x)
#else
#define TANHF(x) tanhf(x)
#endif

// Build a v16bf fragment from two 16-byte LDS chunks (delta in elements).
// elements 0..7 -> VGPRs 0..3 (low-K half), 8..15 -> VGPRs 4..7 (high-K half).
__device__ __forceinline__ v16bf pack2(const bf16_t* p, int delta) {
  v8bf lo = *(const v8bf*)(p);
  v8bf hi = *(const v8bf*)(p + delta);
  v16bf r;
#pragma unroll
  for (int i = 0; i < 8; ++i) { r[i] = lo[i]; r[i + 8] = hi[i]; }
  return r;
}

__device__ __forceinline__ v8f wmma_bf16(v16bf a, v16bf b, v8f c) {
  // D = A(16x32) * B(32x16) + C, f32 accumulate
  return __builtin_amdgcn_wmma_f32_16x16x32_bf16(false, a, false, b, (short)0, c,
                                                 false, false);
}

// Per-wave GEMM slice: (16 x 256) @ (256 x 64) -> 4 C-tiles of 16x16 (f32).
// A streamed from LDS (2 x ds_load_b128 per K-chunk, shared by 4 WMMAs);
// B fragments live permanently in VGPRs.
__device__ __forceinline__ void gemm16r(const bf16_t* __restrict__ inb,
                                        const v16bf wf[4][8],
                                        int lane, v8f acc[4]) {
  const int m  = lane & 15;
  const int hi = lane >> 4;
  const v8f zero = {0.f, 0.f, 0.f, 0.f, 0.f, 0.f, 0.f, 0.f};
#pragma unroll
  for (int t = 0; t < 4; ++t) acc[t] = zero;
  const bf16_t* arow = inb + m * WSTRIDE;
#pragma unroll
  for (int kk = 0; kk < 8; ++kk) {
    // A frag: lane holds row M=lane%16; K = 32*kk + 16*(lane/16) + {0..15}
    v16bf a = pack2(arow + kk * 32 + hi * 16, 8);
#pragma unroll
    for (int t = 0; t < 4; ++t)
      acc[t] = wmma_bf16(a, wf[t][kk], acc[t]);
  }
}

__global__ void __launch_bounds__(128, 1)
dopri5_node_kernel(const float* __restrict__ x0, const float* __restrict__ W1,
                   const float* __restrict__ W2, float* __restrict__ out) {
  extern __shared__ __align__(16) char smem[];
  bf16_t* xsb = (bf16_t*)smem;             // 16  x WSTRIDE  stage input (bf16)
  bf16_t* hb  = xsb + 16 * WSTRIDE;        // 16  x WSTRIDE  tanh(xs@W1) (bf16)
  bf16_t* wt  = hb  + 16 * WSTRIDE;        // 256 x WSTRIDE  staging: W^T [N][K]

  const int tid  = threadIdx.x;
  const int wave = tid >> 5;
  const int lane = tid & 31;
  const int m    = lane & 15;
  const int hi   = lane >> 4;
  const int r0   = blockIdx.x * 16;        // this block's 16 rows of x

  // Stage a 256x256 f32 weight matrix into LDS, transposed [N][K], bf16.
  auto stage_wT = [&](const float* __restrict__ W) {
    for (int idx = tid; idx < 256 * 256; idx += 128) {
      const int n = idx & 255;
      const int kx = idx >> 8;
      wt[n * WSTRIDE + kx] = (bf16_t)W[kx * 256 + n];
    }
  };

  // Pull this wave's 64-column B-fragment slice out of the staging buffer.
  auto load_frags = [&](v16bf wf[4][8]) {
#pragma unroll
    for (int t = 0; t < 4; ++t) {
      const bf16_t* row = wt + (wave * 64 + t * 16 + m) * WSTRIDE + hi * 16;
#pragma unroll
      for (int kk = 0; kk < 8; ++kk)
        wf[t][kk] = pack2(row + kk * 32, 8);   // K = 32*kk + 16*hi + {0..15}
    }
  };

  // --- one-time: both weight matrices into registers (staged via one LDS buffer)
  v16bf w1f[4][8], w2f[4][8];
  stage_wT(W1);
  __syncthreads();
  load_frags(w1f);
  __syncthreads();
  stage_wT(W2);
  __syncthreads();
  load_frags(w2f);

  // Load x tile into WMMA C-layout registers: VGPR v, lane -> M = v + 8*(lane/16),
  // N = 64*wave + 16*t + lane%16. State never leaves registers until the end.
  v8f x[4];
#pragma unroll
  for (int t = 0; t < 4; ++t) {
    const int n = wave * 64 + t * 16 + m;
#pragma unroll
    for (int v = 0; v < 8; ++v)
      x[t][v] = x0[(r0 + v + 8 * hi) * 256 + n];
  }
  __syncthreads();

  // Dormand-Prince a_ij
  const float A1 = 0.2f;
  const float A2[2] = {3.f / 40.f, 9.f / 40.f};
  const float A3[3] = {44.f / 45.f, -56.f / 15.f, 32.f / 9.f};
  const float A4[4] = {19372.f / 6561.f, -25360.f / 2187.f, 64448.f / 6561.f,
                       -212.f / 729.f};
  const float A5[5] = {9017.f / 3168.f, -355.f / 33.f, 46732.f / 5247.f,
                       49.f / 176.f, -5103.f / 18656.f};

  v8f k[6][4];

#pragma unroll 1
  for (int step = 0; step < NSTEPS; ++step) {
#pragma unroll
    for (int s = 0; s < 6; ++s) {
      // xs = x + h * sum_j a_sj k_j  (registers) -> bf16 -> LDS
#pragma unroll
      for (int t = 0; t < 4; ++t) {
        const int n = wave * 64 + t * 16 + m;
#pragma unroll
        for (int v = 0; v < 8; ++v) {
          float val = x[t][v];
          if (s == 1) val += HSTEP * (A1 * k[0][t][v]);
          if (s == 2) val += HSTEP * (A2[0] * k[0][t][v] + A2[1] * k[1][t][v]);
          if (s == 3) val += HSTEP * (A3[0] * k[0][t][v] + A3[1] * k[1][t][v] +
                                      A3[2] * k[2][t][v]);
          if (s == 4) val += HSTEP * (A4[0] * k[0][t][v] + A4[1] * k[1][t][v] +
                                      A4[2] * k[2][t][v] + A4[3] * k[3][t][v]);
          if (s == 5) val += HSTEP * (A5[0] * k[0][t][v] + A5[1] * k[1][t][v] +
                                      A5[2] * k[2][t][v] + A5[3] * k[3][t][v] +
                                      A5[4] * k[4][t][v]);
          xsb[(v + 8 * hi) * WSTRIDE + n] = (bf16_t)val;
        }
      }
      __syncthreads();

      // h = tanh(xs @ W1)
      v8f acc[4];
      gemm16r(xsb, w1f, lane, acc);
#pragma unroll
      for (int t = 0; t < 4; ++t) {
        const int n = wave * 64 + t * 16 + m;
#pragma unroll
        for (int v = 0; v < 8; ++v)
          hb[(v + 8 * hi) * WSTRIDE + n] = (bf16_t)TANHF(acc[t][v]);
      }
      __syncthreads();

      // k_s = h @ W2 (stays in registers)
      gemm16r(hb, w2f, lane, k[s]);
      __syncthreads();
    }

    // 5th-order update (b2 == 0)
#pragma unroll
    for (int t = 0; t < 4; ++t)
#pragma unroll
      for (int v = 0; v < 8; ++v)
        x[t][v] += HSTEP * ((35.f / 384.f) * k[0][t][v] +
                            (500.f / 1113.f) * k[2][t][v] +
                            (125.f / 192.f) * k[3][t][v] +
                            (-2187.f / 6784.f) * k[4][t][v] +
                            (11.f / 84.f) * k[5][t][v]);
  }

  // write x(T) back in f32
#pragma unroll
  for (int t = 0; t < 4; ++t) {
    const int n = wave * 64 + t * 16 + m;
#pragma unroll
    for (int v = 0; v < 8; ++v)
      out[(r0 + v + 8 * hi) * 256 + n] = x[t][v];
  }
}

extern "C" void kernel_launch(void* const* d_in, const int* in_sizes, int n_in,
                              void* d_out, int out_size, void* d_ws, size_t ws_size,
                              hipStream_t stream) {
  (void)in_sizes; (void)n_in; (void)d_ws; (void)ws_size; (void)out_size;
  const float* x0 = (const float*)d_in[0];
  const float* W1 = (const float*)d_in[1];
  const float* W2 = (const float*)d_in[2];
  float* out = (float*)d_out;

  const size_t lds_bytes =
      (size_t)(16 + 16 + 256) * WSTRIDE * sizeof(bf16_t);  // 152,064 B
  hipFuncSetAttribute(reinterpret_cast<const void*>(dopri5_node_kernel),
                      hipFuncAttributeMaxDynamicSharedMemorySize, (int)lds_bytes);

  // 4096 rows / 16 rows per block = 256 blocks; 4 waves (128 threads) each.
  dopri5_node_kernel<<<4096 / 16, 128, lds_bytes, stream>>>(x0, W1, W2, out);
}
